// PSMNet_37623913513328
// MI455X (gfx1250) — matrix operand order repaired and checked by reading, not compiled, not run on hardware
//
#include <hip/hip_runtime.h>
#include <hip/hip_bf16.h>

// ---------------------------------------------------------------------------
// PSMNet-like stereo pipeline for MI455X (gfx1250).
//  - Only batch 0 of the conv tower is needed (sim/cbca/mc3 use [0,0] only).
//  - 3x3 convs (Cin=32/64, Cout=32) run as implicit GEMM on v_wmma_f32_16x16x32_f16.
//  - NHWC f16 features with a 1-pixel zero halo -> unconditional A-tile loads
//    (no exec-mask juggling / zero-fills in the WMMA loop).
//  - Per-layer packed weights staged into LDS once per workgroup via the
//    Tensor Data Mover (tensor_load_to_lds + s_wait_tensorcnt), ds reads after.
// ---------------------------------------------------------------------------

typedef __attribute__((ext_vector_type(16))) _Float16 v16h;
typedef __attribute__((ext_vector_type(8)))  _Float16 v8h;
typedef __attribute__((ext_vector_type(8)))  float    v8f;
typedef unsigned int u32x4 __attribute__((ext_vector_type(4)));
typedef int          i32x8 __attribute__((ext_vector_type(8)));
typedef int          i32x4 __attribute__((ext_vector_type(4)));

#define HQ 192          // quarter-res spatial
#define SP 194          // padded row stride (1-px halo each side)
#define HF 768          // full-res spatial
#define NB 4            // batch

#if defined(__has_builtin)
#  if __has_builtin(__builtin_amdgcn_tensor_load_to_lds) && __has_builtin(__builtin_amdgcn_s_wait_tensorcnt)
#    define HAVE_TDM 1
#  endif
#endif

static __device__ __forceinline__ int iclamp(int v, int lo, int hi) {
    return v < lo ? lo : (v > hi ? hi : v);
}

#ifdef HAVE_TDM
// 1-D TDM copy: global -> LDS, `bytes` must be a multiple of 4 and <= 65535*4.
// D# per CDNA5 ISA: group0 = {count=1, lds_addr, global_addr, type=2},
// group1 = {mask=0, data_size=2 (4B), tensor_dim0=tile_dim0=elems, dim1=1}.
static __device__ __forceinline__ void tdm_load_1d(const void* gptr, void* lptr, unsigned bytes) {
    unsigned elems = bytes >> 2;
    unsigned long long ga = (unsigned long long)(uintptr_t)gptr;
    unsigned lds = (unsigned)(uintptr_t)lptr;        // LDS aperture: addr[31:0] = offset
    u32x4 g0;
    g0[0] = 1u;                                      // count=1, user mode
    g0[1] = lds;                                     // lds_addr
    g0[2] = (unsigned)(ga & 0xffffffffu);            // global_addr[31:0]
    g0[3] = (unsigned)((ga >> 32) & 0x01ffffffu) | (2u << 30);  // addr[56:32] | type=2
    i32x8 g1;
    g1[0] = (int)(2u << 16);                         // workgroup_mask=0, data_size=4B
    g1[1] = (int)((elems & 0xffffu) << 16);          // tensor_dim0[15:0]
    g1[2] = (int)(((elems >> 16) & 0xffffu) | (1u << 16));  // dim0[31:16], tensor_dim1=1
    g1[3] = (int)((elems & 0xffffu) << 16);          // tile_dim0 = elems
    g1[4] = 1;                                       // tile_dim1=1, tile_dim2=0
    g1[5] = (int)elems;                              // tensor_dim0_stride
    g1[6] = 0;
    g1[7] = 0;
    i32x4 z4 = {0, 0, 0, 0};
#if __clang_major__ >= 23
    i32x8 z8 = {0, 0, 0, 0, 0, 0, 0, 0};
    __builtin_amdgcn_tensor_load_to_lds(g0, g1, z4, z4, z8, 0);
#else
    __builtin_amdgcn_tensor_load_to_lds(g0, g1, z4, z4, 0);
#endif
}
#endif

// ---------------------------------------------------------------------------
// Zero-fill (for halo-padded buffers), 4B granules.
// ---------------------------------------------------------------------------
__global__ void __launch_bounds__(256)
zero_u32(unsigned* __restrict__ p, int n) {
    int i = blockIdx.x * blockDim.x + threadIdx.x;
    if (i < n) p[i] = 0u;
}

// ---------------------------------------------------------------------------
// Weight packing: OIHW f32 -> per-lane B-tile register image (f16).
// Layout: wp[((nt*9 + kidx)*KG + kg)*32 + lane][e], e=0..15
//   N = lane&15, K = e + 16*(lane>=16), cout = nt*16+N, cin = kg*32+K
// ---------------------------------------------------------------------------
__global__ void pack_w_kernel(const float* __restrict__ w, _Float16* __restrict__ wp, int KG) {
    int id = blockIdx.x * blockDim.x + threadIdx.x;
    int total = 2 * 9 * KG * 32 * 16;
    if (id >= total) return;
    int e    = id & 15;
    int lane = (id >> 4) & 31;
    int q    = id >> 9;            // (nt*9 + kidx)*KG + kg
    int kg   = q % KG;
    int q2   = q / KG;
    int kidx = q2 % 9;
    int nt   = q2 / 9;
    int N    = lane & 15;
    int K    = e + ((lane >= 16) ? 16 : 0);
    int cout = nt * 16 + N;
    int cin  = kg * 32 + K;
    int Cin  = KG * 32;
    wp[id] = (_Float16)w[((size_t)cout * Cin + cin) * 9 + kidx];
}

// ---------------------------------------------------------------------------
// Feature extraction (b=0 only): conv 3->32, k=3, stride=4, pad=1, ReLU.
// Writes concatenated [left|right] features, NHWC f16, into padded interior.
// ---------------------------------------------------------------------------
__global__ void __launch_bounds__(256)
fe_kernel(const float* __restrict__ left, const float* __restrict__ right,
          const float* __restrict__ few, _Float16* __restrict__ feat) {
    __shared__ float w[32 * 27];
    for (int t = threadIdx.x; t < 32 * 27; t += blockDim.x) w[t] = few[t];
    __syncthreads();
    int idx = blockIdx.x * blockDim.x + threadIdx.x;
    if (idx >= HQ * HQ) return;
    int y = idx / HQ, x = idx % HQ;
    float accL[32], accR[32];
#pragma unroll
    for (int c = 0; c < 32; ++c) { accL[c] = 0.f; accR[c] = 0.f; }
    for (int ci = 0; ci < 3; ++ci)
        for (int ky = 0; ky < 3; ++ky) {
            int iy = 4 * y + ky - 1;
            if ((unsigned)iy >= (unsigned)HF) continue;
            for (int kx = 0; kx < 3; ++kx) {
                int ix = 4 * x + kx - 1;
                if ((unsigned)ix >= (unsigned)HF) continue;
                size_t off = ((size_t)ci * HF + iy) * HF + ix;   // batch 0 NCHW
                float lv = left[off], rv = right[off];
                int wi = ci * 9 + ky * 3 + kx;
#pragma unroll
                for (int c = 0; c < 32; ++c) {
                    float wv = w[c * 27 + wi];
                    accL[c] += wv * lv;
                    accR[c] += wv * rv;
                }
            }
        }
    _Float16* fo = feat + ((size_t)(y + 1) * SP + (x + 1)) * 64;
#pragma unroll
    for (int c = 0; c < 32; ++c) {
        fo[c]      = (_Float16)fmaxf(accL[c], 0.f);
        fo[32 + c] = (_Float16)fmaxf(accR[c], 0.f);
    }
}

// ---------------------------------------------------------------------------
// WMMA implicit-GEMM 3x3 conv over halo-padded NHWC f16, Cin = KG*32, Cout = 32,
// fused affine (+opt ReLU). Weights TDM-staged to LDS once per workgroup.
// One wave = one M-tile (16 consecutive x of one row) x both N-tiles.
// ---------------------------------------------------------------------------
template <int KG, bool RELU>
__global__ void __launch_bounds__(128)
conv3x3_wmma(const _Float16* __restrict__ in,   // [SP][SP][KG*32], zero halo
             const _Float16* __restrict__ wp,   // packed B tiles (global)
             const float* __restrict__ scale, const float* __restrict__ bias,
             _Float16* __restrict__ out) {      // [SP][SP][32], interior written
    const int Cin = KG * 32;
    const int WHALVES = 2 * 9 * KG * 32 * 16;   // packed weight block, halves
    __shared__ _Float16 wlds[WHALVES];

#ifdef HAVE_TDM
    if (threadIdx.x < 32) {                     // wave 0 issues the TDM descriptor
        tdm_load_1d(wp, wlds, WHALVES * 2);
        __builtin_amdgcn_s_wait_tensorcnt(0);
    }
#else
    for (int t = threadIdx.x; t < WHALVES / 8; t += blockDim.x)
        ((v8h*)wlds)[t] = ((const v8h*)wp)[t];
#endif
    __syncthreads();

    int wave = threadIdx.x >> 5;
    int lane = threadIdx.x & 31;
    int tile = blockIdx.x * 4 + wave;           // 0 .. 192*12-1
    int ty = tile / (HQ / 16);
    int tx = (tile % (HQ / 16)) * 16;
    int m  = lane & 15;
    int hs = lane >> 4;                         // half-select (K swizzle)

    v8f acc0 = {};
    v8f acc1 = {};
    union AV { v16h v; v8h h[2]; };

#pragma unroll
    for (int ky = 0; ky < 3; ++ky) {
#pragma unroll
        for (int kx = 0; kx < 3; ++kx) {
            int kidx = ky * 3 + kx;
#pragma unroll
            for (int kg = 0; kg < KG; ++kg) {
                // Padded coords: row ty+ky, col tx+m+kx are always in range.
                const _Float16* rp = in + (((size_t)(ty + ky) * SP + (tx + m + kx)) * Cin
                                           + kg * 32 + hs * 16);
                AV a;
                a.h[0] = *(const v8h*)(rp);
                a.h[1] = *(const v8h*)(rp + 8);
                const _Float16* wb0 = &wlds[((((size_t)0 * 9 + kidx) * KG + kg) * 32 + lane) * 16];
                const _Float16* wb1 = &wlds[((((size_t)1 * 9 + kidx) * KG + kg) * 32 + lane) * 16];
                v16h b0t = *(const v16h*)wb0;
                v16h b1t = *(const v16h*)wb1;
                acc0 = __builtin_amdgcn_wmma_f32_16x16x32_f16(false, a.v, false, b0t,
                                                              (short)0, acc0, false, false);
                acc1 = __builtin_amdgcn_wmma_f32_16x16x32_f16(false, a.v, false, b1t,
                                                              (short)0, acc1, false, false);
            }
        }
    }

    // Epilogue: D VGPR r holds (M = r + 8*hs, N = lane&15).
    int n = lane & 15;
    float s0 = scale[n],      c0 = bias[n];
    float s1 = scale[16 + n], c1 = bias[16 + n];
#pragma unroll
    for (int r = 0; r < 8; ++r) {
        int mm = r + hs * 8;
        size_t o = ((size_t)(ty + 1) * SP + (tx + mm + 1)) * 32;
        float v0 = acc0[r] * s0 + c0;
        float v1 = acc1[r] * s1 + c1;
        if (RELU) { v0 = fmaxf(v0, 0.f); v1 = fmaxf(v1, 0.f); }
        out[o + n]      = (_Float16)v0;
        out[o + 16 + n] = (_Float16)v1;
    }
}

// ---------------------------------------------------------------------------
// Final 32->1 conv + affine + sigmoid -> pred4 (f32, b=0). Reads padded input.
// ---------------------------------------------------------------------------
__global__ void __launch_bounds__(256)
conv6_sigmoid(const _Float16* __restrict__ in, const float* __restrict__ w6,
              const float* __restrict__ s6, const float* __restrict__ b6,
              float* __restrict__ pred) {
    int idx = blockIdx.x * blockDim.x + threadIdx.x;
    if (idx >= HQ * HQ) return;
    int y = idx / HQ, x = idx % HQ;
    float acc = 0.f;
#pragma unroll
    for (int ky = 0; ky < 3; ++ky)
#pragma unroll
        for (int kx = 0; kx < 3; ++kx) {
            const _Float16* rp = in + ((size_t)(y + ky) * SP + (x + kx)) * 32;
            int wb = ky * 3 + kx;
#pragma unroll
            for (int ci = 0; ci < 32; ++ci)
                acc += (float)rp[ci] * w6[ci * 9 + wb];
        }
    float v = acc * s6[0] + b6[0];
    pred[idx] = 1.f / (1.f + expf(-v));
}

// ---------------------------------------------------------------------------
// Bilinear 4x upsample 192->768 (half-pixel centers, edge clamp), b=0.
// ---------------------------------------------------------------------------
__global__ void __launch_bounds__(256)
upsample4(const float* __restrict__ p, float* __restrict__ sim) {
    int idx = blockIdx.x * blockDim.x + threadIdx.x;
    if (idx >= HF * HF) return;
    int oy = idx / HF, ox = idx % HF;
    float sy = (oy + 0.5f) * 0.25f - 0.5f;
    float sx = (ox + 0.5f) * 0.25f - 0.5f;
    int y0 = (int)floorf(sy); float fy = sy - (float)y0;
    int x0 = (int)floorf(sx); float fx = sx - (float)x0;
    int y0c = iclamp(y0, 0, HQ - 1), y1c = iclamp(y0 + 1, 0, HQ - 1);
    int x0c = iclamp(x0, 0, HQ - 1), x1c = iclamp(x0 + 1, 0, HQ - 1);
    float v00 = p[y0c * HQ + x0c], v01 = p[y0c * HQ + x1c];
    float v10 = p[y1c * HQ + x0c], v11 = p[y1c * HQ + x1c];
    sim[idx] = (1.f - fy) * ((1.f - fx) * v00 + fx * v01)
             +         fy * ((1.f - fx) * v10 + fx * v11);
}

// maxpool3 stride3 pad1 -> only the top-left 3x3 outputs are ever consumed.
__global__ void mc3_kernel(const float* __restrict__ sim, float* __restrict__ mc3) {
    int t = threadIdx.x;
    if (t >= 9) return;
    int i = t / 3, j = t % 3;
    float m = -__builtin_inff();
    for (int dy = -1; dy <= 1; ++dy) {
        int y = 3 * i + dy;
        if ((unsigned)y >= (unsigned)HF) continue;
        for (int dx = -1; dx <= 1; ++dx) {
            int x = 3 * j + dx;
            if ((unsigned)x >= (unsigned)HF) continue;
            m = fmaxf(m, sim[y * HF + x]);
        }
    }
    mc3[t] = m;
}

// avgpool3 (zero pad, always /9); optional fused subtraction (cbca = pool^4 - sim).
__global__ void __launch_bounds__(256)
avgpool3_k(const float* __restrict__ src, float* __restrict__ dst,
           const float* __restrict__ sub) {
    int idx = blockIdx.x * blockDim.x + threadIdx.x;
    if (idx >= HF * HF) return;
    int y = idx / HF, x = idx % HF;
    float s = 0.f;
#pragma unroll
    for (int dy = -1; dy <= 1; ++dy) {
        int yy = y + dy;
        if ((unsigned)yy >= (unsigned)HF) continue;
#pragma unroll
        for (int dx = -1; dx <= 1; ++dx) {
            int xx = x + dx;
            if ((unsigned)xx >= (unsigned)HF) continue;
            s += src[yy * HF + xx];
        }
    }
    s *= (1.f / 9.f);
    if (sub) s -= sub[idx];
    dst[idx] = s;
}

// batched avgpool3 of Csgm -> final output [B][768][768]
__global__ void __launch_bounds__(256)
avgpool3_batched(const float* __restrict__ src, float* __restrict__ dst) {
    int idx = blockIdx.x * blockDim.x + threadIdx.x;
    if (idx >= HF * HF) return;
    int b = blockIdx.y;
    const float* s0 = src + (size_t)b * HF * HF;
    float* d0 = dst + (size_t)b * HF * HF;
    int y = idx / HF, x = idx % HF;
    float s = 0.f;
#pragma unroll
    for (int dy = -1; dy <= 1; ++dy) {
        int yy = y + dy;
        if ((unsigned)yy >= (unsigned)HF) continue;
#pragma unroll
        for (int dx = -1; dx <= 1; ++dx) {
            int xx = x + dx;
            if ((unsigned)xx >= (unsigned)HF) continue;
            s += s0[yy * HF + xx];
        }
    }
    d0[idx] = s * (1.f / 9.f);
}

// ---------------------------------------------------------------------------
// Bug-faithful SGM cost (per batch; sim2/cbca/mc3 are batch-0 maps).
// ---------------------------------------------------------------------------
__global__ void __launch_bounds__(256)
sgm_kernel(const float* __restrict__ left, const float* __restrict__ right,
           const float* __restrict__ sim2, const float* __restrict__ cbca,
           const float* __restrict__ mc3, float* __restrict__ csgm) {
    const int H = HF, W = HF;
    int idx = blockIdx.x * blockDim.x + threadIdx.x;
    if (idx >= H * W) return;
    int b = blockIdx.y;
    int i = idx / W, j = idx % W;

    int r_j = (j > W - 3) ? (W - 1) : 0;
    int t_i = (j >= 2) ? j : ((i < 2) ? 0 : i);                       // bug-faithful
    int b_i = (j <= W - 3) ? j : ((i > H - 3) ? (H - 1) : i);

    const size_t chs = (size_t)H * W;
    const float* Lb = left  + (size_t)b * 3 * chs;
    const float* Rb = right + (size_t)b * 3 * chs;
    size_t p  = (size_t)i * W + j;
    size_t pl = (size_t)i * W;
    size_t pr = (size_t)i * W + r_j;
    size_t pt = (size_t)t_i * W + j;
    size_t pb = (size_t)b_i * W + j;

    float ld1 = 0, ld2 = 0, rd1 = 0, rd2 = 0, td1 = 0, td2 = 0, bd1 = 0, bd2 = 0;
#pragma unroll
    for (int c = 0; c < 3; ++c) {
        size_t cs = (size_t)c * chs;
        float lc = Lb[cs + p], rc = Rb[cs + p];
        ld1 += fabsf(lc - Lb[cs + pl]); ld2 += fabsf(rc - Rb[cs + pl]);
        rd1 += fabsf(lc - Lb[cs + pr]); rd2 += fabsf(rc - Rb[cs + pr]);
        td1 += fabsf(lc - Lb[cs + pt]); td2 += fabsf(rc - Rb[cs + pt]);
        bd1 += fabsf(lc - Lb[cs + pb]); bd2 += fabsf(rc - Rb[cs + pb]);
    }
    const float inv3 = 1.f / 3.f;
    ld1 *= inv3; ld2 *= inv3; rd1 *= inv3; rd2 *= inv3;
    td1 *= inv3; td2 *= inv3; bd1 *= inv3; bd2 *= inv3;

    float maxc = mc3[(i % 3) * 3 + (j % 3)];
    float cbp  = cbca[p];
    float cost_l = sim2[pl], mc_l = mc3[(i % 3) * 3 + 0];
    float cost_r = sim2[pr], mc_r = mc3[(i % 3) * 3 + (r_j % 3)];
    float cost_t = sim2[pt], mc_t = mc3[(t_i % 3) * 3 + (j % 3)];
    float cost_b = sim2[pb], mc_b = mc3[(b_i % 3) * 3 + (j % 3)];

    auto csg = [&](float cpr, float mpr, float d1, float d2) -> float {
        const float D = 0.08f;
        bool cond = ((d1 > D) && (d2 > D)) || ((d1 < D) && (d2 < D));
        float p1 = cond ? 0.24f : 0.32f;
        float p2 = cond ? 0.56f : 1.28f;
        float a = maxc, b2 = mpr + p1, c2 = cpr + p2;
        float mn = fminf(fminf(a, b2), c2);
        float mx = fmaxf(fmaxf(a, b2), c2);
        return cbp + (mn + mx) * 0.5f;
    };

    float out = (csg(cost_l, mc_l, ld1, ld2) + csg(cost_r, mc_r, rd1, rd2)
               + csg(cost_b, mc_b, bd1, bd2) + csg(cost_t, mc_t, td1, td2)) * 0.25f;
    csgm[(size_t)b * chs + p] = out;
}

// ---------------------------------------------------------------------------
extern "C" void kernel_launch(void* const* d_in, const int* in_sizes, int n_in,
                              void* d_out, int out_size, void* d_ws, size_t ws_size,
                              hipStream_t stream) {
    const float* left  = (const float*)d_in[0];
    const float* right = (const float*)d_in[1];
    const float* fe_w  = (const float*)d_in[2];
    const float* w1 = (const float*)d_in[3];
    const float* s1 = (const float*)d_in[4];
    const float* b1 = (const float*)d_in[5];
    const float* w2 = (const float*)d_in[6];
    const float* s2 = (const float*)d_in[7];
    const float* b2 = (const float*)d_in[8];
    const float* w3 = (const float*)d_in[9];
    const float* s3 = (const float*)d_in[10];
    const float* b3 = (const float*)d_in[11];
    const float* w4 = (const float*)d_in[12];
    const float* s4 = (const float*)d_in[13];
    const float* b4 = (const float*)d_in[14];
    const float* w5 = (const float*)d_in[15];
    const float* s5 = (const float*)d_in[16];
    const float* b5 = (const float*)d_in[17];
    const float* w6 = (const float*)d_in[18];
    const float* s6 = (const float*)d_in[19];
    const float* b6 = (const float*)d_in[20];

    // ---- workspace carve (256B aligned) ----
    char* ws = (char*)d_ws;
    size_t off = 0;
    auto carve = [&](size_t bytes) -> void* {
        void* p = ws + off;
        off += (bytes + 255) & ~(size_t)255;
        return p;
    };
    const size_t featB = (size_t)SP * SP * 64 * 2;   // padded, 4.82 MB
    const size_t bufB  = (size_t)SP * SP * 32 * 2;   // padded, 2.41 MB
    _Float16* feat  = (_Float16*)carve(featB);
    _Float16* buf0  = (_Float16*)carve(bufB);
    _Float16* buf1  = (_Float16*)carve(bufB);
    _Float16* wp1   = (_Float16*)carve((size_t)2 * 9 * 2 * 32 * 16 * 2);
    _Float16* wp2   = (_Float16*)carve((size_t)2 * 9 * 1 * 32 * 16 * 2);
    _Float16* wp3   = (_Float16*)carve((size_t)2 * 9 * 1 * 32 * 16 * 2);
    _Float16* wp4   = (_Float16*)carve((size_t)2 * 9 * 1 * 32 * 16 * 2);
    _Float16* wp5   = (_Float16*)carve((size_t)2 * 9 * 1 * 32 * 16 * 2);
    float* pred4    = (float*)carve((size_t)HQ * HQ * 4);
    float* sim      = (float*)carve((size_t)HF * HF * 4);
    float* pA       = (float*)carve((size_t)HF * HF * 4);
    float* pB       = (float*)carve((size_t)HF * HF * 4);
    float* mc3      = (float*)carve(64);
    float* csgm     = (float*)carve((size_t)NB * HF * HF * 4);      // 9.4 MB
    float* outp     = (float*)d_out;                                // [4][768][768] f32

    // ---- zero halo-padded buffers (deterministic each call) ----
    {
        int nf = (int)(featB / 4), nb = (int)(bufB / 4);
        zero_u32<<<(nf + 255) / 256, 256, 0, stream>>>((unsigned*)feat, nf);
        zero_u32<<<(nb + 255) / 256, 256, 0, stream>>>((unsigned*)buf0, nb);
        zero_u32<<<(nb + 255) / 256, 256, 0, stream>>>((unsigned*)buf1, nb);
    }

    // ---- weight packing ----
    pack_w_kernel<<<(2 * 9 * 2 * 512 + 255) / 256, 256, 0, stream>>>(w1, wp1, 2);
    pack_w_kernel<<<(2 * 9 * 1 * 512 + 255) / 256, 256, 0, stream>>>(w2, wp2, 1);
    pack_w_kernel<<<(2 * 9 * 1 * 512 + 255) / 256, 256, 0, stream>>>(w3, wp3, 1);
    pack_w_kernel<<<(2 * 9 * 1 * 512 + 255) / 256, 256, 0, stream>>>(w4, wp4, 1);
    pack_w_kernel<<<(2 * 9 * 1 * 512 + 255) / 256, 256, 0, stream>>>(w5, wp5, 1);

    // ---- feature extraction (b=0 only) ----
    fe_kernel<<<(HQ * HQ + 255) / 256, 256, 0, stream>>>(left, right, fe_w, feat);

    // ---- WMMA conv tower ----
    int convBlocks = (HQ * (HQ / 16)) / 4;   // 2304 M-tiles / 4 waves per block
    conv3x3_wmma<2, true ><<<convBlocks, 128, 0, stream>>>(feat, wp1, s1, b1, buf0);
    conv3x3_wmma<1, true ><<<convBlocks, 128, 0, stream>>>(buf0, wp2, s2, b2, buf1);
    conv3x3_wmma<1, true ><<<convBlocks, 128, 0, stream>>>(buf1, wp3, s3, b3, buf0);
    conv3x3_wmma<1, false><<<convBlocks, 128, 0, stream>>>(buf0, wp4, s4, b4, buf1); // c0
    conv3x3_wmma<1, true ><<<convBlocks, 128, 0, stream>>>(buf1, wp5, s5, b5, buf0);

    // ---- 32->1 conv + sigmoid, upsample, mc3 ----
    conv6_sigmoid<<<(HQ * HQ + 255) / 256, 256, 0, stream>>>(buf0, w6, s6, b6, pred4);
    upsample4<<<(HF * HF + 255) / 256, 256, 0, stream>>>(pred4, sim);
    mc3_kernel<<<1, 32, 0, stream>>>(sim, mc3);

    // ---- 4x avgpool chain; last one fuses cbca = pool^4(sim) - sim ----
    int pg = (HF * HF + 255) / 256;
    avgpool3_k<<<pg, 256, 0, stream>>>(sim, pA, (const float*)nullptr);
    avgpool3_k<<<pg, 256, 0, stream>>>(pA, pB, (const float*)nullptr);
    avgpool3_k<<<pg, 256, 0, stream>>>(pB, pA, (const float*)nullptr);
    avgpool3_k<<<pg, 256, 0, stream>>>(pA, pB, sim);                 // pB = cbca4[0,0]

    // ---- SGM (per batch) + final avgpool ----
    dim3 sg((HF * HF + 255) / 256, NB, 1);
    sgm_kernel<<<sg, 256, 0, stream>>>(left, right, sim, pB, mc3, csgm);
    avgpool3_batched<<<sg, 256, 0, stream>>>(csgm, outp);
    (void)in_sizes; (void)n_in; (void)out_size; (void)ws_size;
}
